// FCAF3D_26620207301334
// MI455X (gfx1250) — matrix operand order repaired and checked by reading, not compiled, not run on hardware
//
#include <hip/hip_runtime.h>
#include <cstdint>

// ============================================================================
// FCAF3D / PointNet++ forward for MI455X (gfx1250, wave32, WMMA).
//
// d_in layout assumption (jax tree_leaves: dict keys sorted 'fp'<'heads'<'sa',
// 'W'<'b'):
//   [0]                points (B,8192,6) f32
//   [1..16]            fp:   4 levels x 2 layers x (W,b)
//   [17..40]           heads: 3 heads x (cls,ctr,off,size) x (W,b)
//   [41..64]           sa:   4 levels x 3 layers x (W,b)
//
// All MLP GEMMs run as f16 x f16 -> f32-accum WMMA (v_wmma_f32_16x16x32_f16),
// fused bias+ReLU epilogue, f16 activations. Every GEMM's padded N is a
// multiple of 64 (the block N tile), so the WMMA inner loop has no guards and
// EXEC stays all-ones with zero saveexec overhead. Workspace use ~160 MB.
// ============================================================================

typedef __attribute__((ext_vector_type(16))) _Float16 v16h;
typedef __attribute__((ext_vector_type(8)))  _Float16 v8h;
typedef __attribute__((ext_vector_type(8)))  float    v8f;

#define CEILDIV(a, b) (((a) + (b) - 1) / (b))
static inline int align32i(int x) { return (x + 31) & ~31; }
static inline int align64i(int x) { return (x + 63) & ~63; }

// ------------------------------------------------------------------ utilities

__global__ void split_points_kernel(const float* __restrict__ pts,
                                    float* __restrict__ xyz,
                                    _Float16* __restrict__ feats, int total) {
  int e = blockIdx.x * blockDim.x + threadIdx.x;
  if (e >= total) return;
  int i = e / 6, c = e % 6;
  float v = pts[e];
  if (c < 3) xyz[i * 3 + c] = v;
  else       feats[i * 3 + (c - 3)] = (_Float16)v;
}

// Pad weights (cout x cin, f32) to (Npad x Kpad, f16), bias to Npad f32.
__global__ void convert_weights_kernel(const float* __restrict__ W,
                                       const float* __restrict__ bias,
                                       _Float16* __restrict__ Wp,
                                       float* __restrict__ bp,
                                       int cout, int cin, int Npad, int Kpad) {
  int e = blockIdx.x * blockDim.x + threadIdx.x;
  int total = Npad * Kpad;
  if (e < total) {
    int r = e / Kpad, c = e % Kpad;
    Wp[e] = (r < cout && c < cin) ? (_Float16)W[r * cin + c] : (_Float16)0.0f;
  }
  if (e < Npad) bp[e] = (e < cout) ? bias[e] : 0.0f;
}

// -------------------------------------------------------------- FPS (in LDS)
// One workgroup per batch. LDS: x[N], y[N], z[N], dist[N], red arrays.
__global__ void fps_kernel(const float* __restrict__ xyz, int Npts, int npoint,
                           int* __restrict__ out_idx) {
  extern __shared__ float sm[];
  float* sx = sm;
  float* sy = sm + Npts;
  float* sz = sm + 2 * Npts;
  float* dist = sm + 3 * Npts;
  float* rval = sm + 4 * Npts;
  int*   ridx = (int*)(rval + blockDim.x);

  const float* xb = xyz + (size_t)blockIdx.x * Npts * 3;
  int* ob = out_idx + (size_t)blockIdx.x * npoint;
  int tid = threadIdx.x, bdim = blockDim.x;

  for (int p = tid; p < Npts; p += bdim) {
    sx[p] = xb[p * 3 + 0];
    sy[p] = xb[p * 3 + 1];
    sz[p] = xb[p * 3 + 2];
    dist[p] = 1e10f;
  }
  __syncthreads();

  int far = 0;
  for (int j = 0; j < npoint; ++j) {
    if (tid == 0) ob[j] = far;
    float fx = sx[far], fy = sy[far], fz = sz[far];
    float best = -1.0f;
    int bi = 0;
    for (int p = tid; p < Npts; p += bdim) {
      float dx = sx[p] - fx, dy = sy[p] - fy, dz = sz[p] - fz;
      float d = dx * dx + dy * dy + dz * dz;
      float dd = dist[p];
      dd = dd < d ? dd : d;
      dist[p] = dd;
      if (dd > best) { best = dd; bi = p; }
    }
    rval[tid] = best;
    ridx[tid] = bi;
    __syncthreads();
    for (int s = bdim >> 1; s > 0; s >>= 1) {
      if (tid < s) {
        float v2 = rval[tid + s];
        int   i2 = ridx[tid + s];
        if (v2 > rval[tid] || (v2 == rval[tid] && i2 < ridx[tid])) {
          rval[tid] = v2;
          ridx[tid] = i2;
        }
      }
      __syncthreads();
    }
    far = ridx[0];
    __syncthreads();
  }
}

__global__ void gather_points_kernel(const float* __restrict__ src,
                                     const int* __restrict__ idx,
                                     float* __restrict__ dst,
                                     int Nsrc, int np, int totalPts) {
  int e = blockIdx.x * blockDim.x + threadIdx.x;
  if (e >= totalPts) return;                // totalPts = B*np
  int b = e / np, i = e % np;
  int id = idx[(size_t)b * np + i];
  const float* s = src + ((size_t)b * Nsrc + id) * 3;
  float* d = dst + (size_t)e * 3;
  d[0] = s[0]; d[1] = s[1]; d[2] = s[2];
}

// ----------------------------------------------------------- ball query
// One wave per center; collect first 32 in-radius indices in ascending order,
// pad with the first hit (reference semantics: sort, take 32, pad w/ first).
__global__ void ball_query_kernel(const float* __restrict__ xyz,
                                  const float* __restrict__ newxyz,
                                  int* __restrict__ gi,
                                  int Npts, int np, float r2, int totalC) {
  __shared__ int sbuf[4][32];
  int lane = threadIdx.x & 31;
  int wv = threadIdx.x >> 5;
  int center = blockIdx.x * 4 + wv;
  bool active = center < totalC;
  float cx = 0, cy = 0, cz = 0;
  const float* xb = xyz;
  if (active) {
    int b = center / np;
    const float* c = newxyz + (size_t)center * 3;
    cx = c[0]; cy = c[1]; cz = c[2];
    xb = xyz + (size_t)b * Npts * 3;
  }
  int count = 0;
  if (active) {
    for (int base = 0; base < Npts; base += 32) {
      int p = base + lane;
      float dx = xb[p * 3 + 0] - cx;
      float dy = xb[p * 3 + 1] - cy;
      float dz = xb[p * 3 + 2] - cz;
      bool pred = (dx * dx + dy * dy + dz * dz) <= r2;
      unsigned m = (unsigned)__ballot(pred);
      int pre = __popc(m & ((1u << lane) - 1u));
      if (pred && (count + pre) < 32) sbuf[wv][count + pre] = p;
      count += __popc(m);
      if (count >= 32) break;
    }
  }
  __syncthreads();
  if (active) {
    int total = count < 32 ? count : 32;
    if (total == 0) { sbuf[wv][0] = 0; total = 1; }   // cannot happen (self hit)
    int first = sbuf[wv][0];
    int v = (lane < total) ? sbuf[wv][lane] : first;
    gi[((size_t)center) * 32 + lane] = v;
  }
}

// -------------------------------------------------- SA grouping (per batch)
// Builds A-matrix rows: [rel_xyz(3) | feats(Cfeat) | zero-pad] as f16.
__global__ void sa_group_kernel(const float* __restrict__ xyzb,
                                const float* __restrict__ newxyzb,
                                const _Float16* __restrict__ featsb, int Cfeat,
                                const int* __restrict__ gib,
                                _Float16* __restrict__ G,
                                int Kpad, int total) {
  int e = blockIdx.x * blockDim.x + threadIdx.x;
  if (e >= total) return;
  int c = e % Kpad;
  int rs = e / Kpad;
  int s = rs & 31;
  int i = rs >> 5;
  int src = gib[(size_t)i * 32 + s];
  float v = 0.0f;
  if (c < 3)                v = xyzb[src * 3 + c] - newxyzb[i * 3 + c];
  else if (c - 3 < Cfeat)   v = (float)featsb[(size_t)src * Cfeat + (c - 3)];
  G[e] = (_Float16)v;
}

// ----------------------------------------------------------- WMMA GEMM
// C[M x Npad] = relu(A[M x K] * Wp[Npad x K]^T + bp), strides K / Npad.
// M % 128 == 0, K % 32 == 0, Npad % 64 == 0 -> no guards anywhere; EXEC stays
// all-ones around every WMMA (ISA 7.12 requirement) with no saveexec churn.
// Fragment layouts per CDNA5 ISA 7.12.2:
//  A 16x32 f16: lane<16 holds row (lane), K in {0..7}U{16..23}; lane>=16: +8.
//  B 32x16 f16: lane&15 = column, K order same as A ({0..7}U{16..23}; lane>=16: +8).
//  D v8f: m = vr + 8*(lane>=16), n = lane&15.
__device__ __forceinline__ v16h load_fragA(const _Float16* tile, int ld, int lane) {
  const _Float16* p = tile + (size_t)(lane & 15) * ld + ((lane & 16) ? 8 : 0);
  v8h lo = *(const v8h*)p;
  v8h hi = *(const v8h*)(p + 16);
  v16h r;
#pragma unroll
  for (int h = 0; h < 8; ++h) { r[h] = lo[h]; r[h + 8] = hi[h]; }
  return r;
}
__device__ __forceinline__ v16h load_fragB(const _Float16* tile, int ld, int lane) {
  const _Float16* p = tile + (size_t)(lane & 15) * ld + ((lane & 16) ? 8 : 0);
  v8h lo = *(const v8h*)p;
  v8h hi = *(const v8h*)(p + 16);
  v16h r;
#pragma unroll
  for (int h = 0; h < 8; ++h) { r[h] = lo[h]; r[h + 8] = hi[h]; }
  return r;
}

__global__ __launch_bounds__(256)
void wmma_gemm_bias_relu(const _Float16* __restrict__ A,
                         const _Float16* __restrict__ Wp,
                         const float* __restrict__ bp,
                         _Float16* __restrict__ C,
                         int M, int K, int Npad) {
  const int lane = threadIdx.x & 31;
  const int wave = threadIdx.x >> 5;
  const int m0 = blockIdx.x * 128 + (wave & 3) * 32;
  const int n0 = blockIdx.y * 64 + (wave >> 2) * 32;

  v8f acc00 = {}, acc01 = {}, acc10 = {}, acc11 = {};
  const _Float16* A0 = A + (size_t)m0 * K;
  const _Float16* A1 = A + (size_t)(m0 + 16) * K;
  const _Float16* B0 = Wp + (size_t)n0 * K;
  const _Float16* B1 = Wp + (size_t)(n0 + 16) * K;

  for (int k0 = 0; k0 < K; k0 += 32) {
    v16h a0 = load_fragA(A0 + k0, K, lane);
    v16h a1 = load_fragA(A1 + k0, K, lane);
    v16h b0 = load_fragB(B0 + k0, K, lane);
    v16h b1 = load_fragB(B1 + k0, K, lane);
    acc00 = __builtin_amdgcn_wmma_f32_16x16x32_f16(false, a0, false, b0,
                                                   (short)0, acc00, false, false);
    acc10 = __builtin_amdgcn_wmma_f32_16x16x32_f16(false, a1, false, b0,
                                                   (short)0, acc10, false, false);
    acc01 = __builtin_amdgcn_wmma_f32_16x16x32_f16(false, a0, false, b1,
                                                   (short)0, acc01, false, false);
    acc11 = __builtin_amdgcn_wmma_f32_16x16x32_f16(false, a1, false, b1,
                                                   (short)0, acc11, false, false);
  }

  const int colL = lane & 15;
  const int rowAdd = (lane & 16) ? 8 : 0;
  auto store_tile = [&](const v8f& acc, int moff, int noff) {
    int col = n0 + noff + colL;
    float bias = bp[col];
    int rbase = m0 + moff + rowAdd;
#pragma unroll
    for (int vr = 0; vr < 8; ++vr) {
      float v = acc[vr] + bias;
      v = v > 0.0f ? v : 0.0f;
      C[(size_t)(rbase + vr) * Npad + col] = (_Float16)v;
    }
  };
  store_tile(acc00, 0, 0);
  store_tile(acc01, 0, 16);
  store_tile(acc10, 16, 0);
  store_tile(acc11, 16, 16);
}

// ------------------------------------------------------- max over 32 samples
__global__ void maxpool32_kernel(const _Float16* __restrict__ A3,
                                 _Float16* __restrict__ out, int C, int total) {
  int e = blockIdx.x * blockDim.x + threadIdx.x;
  if (e >= total) return;                    // total = np*C (per batch)
  int i = e / C, c = e % C;
  const _Float16* p = A3 + ((size_t)i * 32) * C + c;
  float m = (float)p[0];
#pragma unroll 4
  for (int s = 1; s < 32; ++s) {
    float v = (float)p[(size_t)s * C];
    m = v > m ? v : m;
  }
  out[(size_t)i * C + c] = (_Float16)m;
}

// ------------------------------------------------------- FP: 3-NN + weights
__global__ void three_nn_kernel(const float* __restrict__ xyz1,
                                const float* __restrict__ xyz2,
                                int* __restrict__ nnidx, float* __restrict__ nnw,
                                int N1, int N2, int total) {
  int e = blockIdx.x * blockDim.x + threadIdx.x;
  if (e >= total) return;                    // total = B*N1
  int b = e / N1;
  const float* p = xyz1 + (size_t)e * 3;
  float px = p[0], py = p[1], pz = p[2];
  const float* q = xyz2 + (size_t)b * N2 * 3;
  float d0 = 3e38f, d1 = 3e38f, d2 = 3e38f;
  int j0 = 0, j1 = 0, j2 = 0;
  for (int j = 0; j < N2; ++j) {
    float dx = q[j * 3 + 0] - px, dy = q[j * 3 + 1] - py, dz = q[j * 3 + 2] - pz;
    float d = dx * dx + dy * dy + dz * dz;
    if (d < d0)      { d2 = d1; j2 = j1; d1 = d0; j1 = j0; d0 = d; j0 = j; }
    else if (d < d1) { d2 = d1; j2 = j1; d1 = d; j1 = j; }
    else if (d < d2) { d2 = d; j2 = j; }
  }
  float w0 = 1.0f / (d0 + 1e-8f);
  float w1 = 1.0f / (d1 + 1e-8f);
  float w2 = 1.0f / (d2 + 1e-8f);
  float s = 1.0f / (w0 + w1 + w2);
  nnidx[(size_t)e * 3 + 0] = j0;
  nnidx[(size_t)e * 3 + 1] = j1;
  nnidx[(size_t)e * 3 + 2] = j2;
  nnw[(size_t)e * 3 + 0] = w0 * s;
  nnw[(size_t)e * 3 + 1] = w1 * s;
  nnw[(size_t)e * 3 + 2] = w2 * s;
}

// Build FP GEMM input rows: [skip f1(C1) | interp(C2) | zero-pad] as f16.
__global__ void fp_build_kernel(const _Float16* __restrict__ f1, int C1,
                                const _Float16* __restrict__ f2, int C2,
                                const int* __restrict__ nnidx,
                                const float* __restrict__ nnw,
                                _Float16* __restrict__ G,
                                int N1, int N2, int Kpad, int total) {
  int e = blockIdx.x * blockDim.x + threadIdx.x;
  if (e >= total) return;                    // total = B*N1*Kpad
  int c = e % Kpad;
  int r = e / Kpad;                          // b*N1 + i
  float v = 0.0f;
  if (c < C1) {
    v = (float)f1[(size_t)r * C1 + c];
  } else if (c < C1 + C2) {
    int cc = c - C1;
    int b = r / N1;
    const int* id = nnidx + (size_t)r * 3;
    const float* w = nnw + (size_t)r * 3;
    size_t base = (size_t)b * N2 * C2;
    v = w[0] * (float)f2[base + (size_t)id[0] * C2 + cc]
      + w[1] * (float)f2[base + (size_t)id[1] * C2 + cc]
      + w[2] * (float)f2[base + (size_t)id[2] * C2 + cc];
  }
  G[e] = (_Float16)v;
}

// ------------------------------------------------------------------- heads
// 26 output channels per point (cls19|ctr1|off3|size3), written transposed.
__global__ void head_kernel(const _Float16* __restrict__ f,
                            const float* __restrict__ Wc, const float* __restrict__ bc,
                            const float* __restrict__ Wt, const float* __restrict__ bt,
                            const float* __restrict__ Wo, const float* __restrict__ bo,
                            const float* __restrict__ Ws, const float* __restrict__ bs,
                            float* __restrict__ out, int Nl, int total) {
  int e = blockIdx.x * blockDim.x + threadIdx.x;
  if (e >= total) return;                    // total = B*Nl*26
  int d = e / (4 * Nl);
  int r = e % (4 * Nl);
  int b = r / Nl, n = r % Nl;
  const float* W; const float* bias; float* base; int dg, Dg;
  if (d < 19)      { W = Wc; bias = bc; dg = d;      Dg = 19; base = out; }
  else if (d == 19){ W = Wt; bias = bt; dg = 0;      Dg = 1;  base = out + (size_t)4 * 19 * Nl; }
  else if (d < 23) { W = Wo; bias = bo; dg = d - 20; Dg = 3;  base = out + (size_t)4 * 20 * Nl; }
  else             { W = Ws; bias = bs; dg = d - 23; Dg = 3;  base = out + (size_t)4 * 23 * Nl; }
  const _Float16* fr = f + ((size_t)b * Nl + n) * 256;
  const float* wr = W + (size_t)dg * 256;
  float acc = bias[dg];
  for (int c = 0; c < 256; ++c) acc += (float)fr[c] * wr[c];
  base[((size_t)b * Dg + dg) * Nl + n] = acc;
}

// ============================================================================
// Host orchestration
// ============================================================================
extern "C" void kernel_launch(void* const* d_in, const int* in_sizes, int n_in,
                              void* d_out, int out_size, void* d_ws, size_t ws_size,
                              hipStream_t stream) {
  (void)in_sizes; (void)n_in; (void)out_size; (void)ws_size;
  const int B = 4;
  static const int NPTS[5]   = {8192, 4096, 2048, 1024, 512};
  static const int NPOINT[4] = {4096, 2048, 1024, 512};
  static const float RAD2[4] = {0.01f, 0.04f, 0.16f, 0.64f};
  static const int SAC[4][3] = {{64,64,128},{128,128,256},{256,256,512},{512,512,1024}};
  static const int CSA[5]    = {3, 128, 256, 512, 1024};   // sa_feats channels
  static const int FPC[4][2] = {{512,256},{256,256},{256,256},{259,256}};
  static const int FPCIN[4]  = {1536, 512, 384, 259};

  const float* points = (const float*)d_in[0];
  auto FPW = [](int l, int j) { return 1 + l * 4 + j * 2; };
  auto SAW = [](int l, int j) { return 41 + l * 6 + j * 2; };
  const int HEADB = 17;

  // ---- workspace bump allocator (needs ~160 MB) ----
  char* ws = (char*)d_ws;
  size_t off = 0;
  auto alloc = [&](size_t bytes) -> char* {
    char* p = ws + off;
    off = (off + bytes + 255) & ~(size_t)255;
    return p;
  };

  float* xyz[5];
  for (int l = 0; l < 5; ++l) xyz[l] = (float*)alloc((size_t)B * NPTS[l] * 3 * 4);
  _Float16* safeat[5];
  safeat[0] = (_Float16*)alloc((size_t)B * NPTS[0] * 3 * 2);          // feats0
  for (int l = 1; l < 5; ++l)
    safeat[l] = (_Float16*)alloc((size_t)B * NPTS[l] * CSA[l] * 2);
  _Float16* fpout[4];
  for (int i = 0; i < 4; ++i)
    fpout[i] = (_Float16*)alloc((size_t)B * NPTS[3 - i] * 256 * 2);
  int* fpsidx = (int*)alloc((size_t)B * 4096 * 4);
  int* gi     = (int*)alloc((size_t)B * 4096 * 32 * 4);
  int*   nnidx = (int*)alloc((size_t)B * 8192 * 3 * 4);
  float* nnw   = (float*)alloc((size_t)B * 8192 * 3 * 4);
  _Float16* G  = (_Float16*)alloc((size_t)10485760 * 2);   // max 65536*160
  _Float16* A1 = (_Float16*)alloc((size_t)10485760 * 2);   // max 32768*256
  _Float16* A2 = (_Float16*)alloc((size_t)8388608 * 2);
  _Float16* A3 = (_Float16*)alloc((size_t)16777216 * 2);   // max 131072*128

  // ---- weight layer table (12 SA + 8 FP) ----
  // Kpad: first layer of a chain = align32(cin); chained layers inherit the
  // producer's Npad. Npad: always align64(cout) so GEMM N tiles are exact.
  int wcout[20], wcin[20], wkp[20], wnp[20], widx[20];
  for (int k = 0; k < 12; ++k) {
    int l = k / 3, j = k % 3;
    wcin[k] = (j == 0) ? (3 + CSA[l]) : SAC[l][j - 1];
    wcout[k] = SAC[l][j];
    wnp[k] = align64i(wcout[k]);
    wkp[k] = (j == 0) ? align32i(wcin[k]) : wnp[k - 1];
    widx[k] = SAW(l, j);
  }
  for (int k = 0; k < 8; ++k) {
    int i = k / 2, j = k % 2;
    wcin[12 + k] = (j == 0) ? FPCIN[i] : FPC[i][0];
    wcout[12 + k] = FPC[i][j];
    wnp[12 + k] = align64i(wcout[12 + k]);
    wkp[12 + k] = (j == 0) ? align32i(wcin[12 + k]) : wnp[12 + k - 1];
    widx[12 + k] = FPW(i, j);
  }
  _Float16* wps[20]; float* bps[20];
  for (int k = 0; k < 20; ++k) {
    wps[k] = (_Float16*)alloc((size_t)wnp[k] * wkp[k] * 2);
    bps[k] = (float*)alloc((size_t)wnp[k] * 4);
  }

  // ---- 0) split points, convert weights ----
  {
    int total = B * NPTS[0] * 6;
    split_points_kernel<<<CEILDIV(total, 256), 256, 0, stream>>>(
        points, xyz[0], safeat[0], total);
  }
  for (int k = 0; k < 20; ++k) {
    int total = wnp[k] * wkp[k];
    convert_weights_kernel<<<CEILDIV(total, 256), 256, 0, stream>>>(
        (const float*)d_in[widx[k]], (const float*)d_in[widx[k] + 1],
        wps[k], bps[k], wcout[k], wcin[k], wnp[k], wkp[k]);
  }

  auto launch_gemm = [&](const _Float16* Ain, _Float16* Cout, int layer,
                         int M) {
    dim3 g((unsigned)(M / 128), (unsigned)(wnp[layer] / 64));
    wmma_gemm_bias_relu<<<g, 256, 0, stream>>>(Ain, wps[layer], bps[layer],
                                               Cout, M, wkp[layer], wnp[layer]);
  };

  // ---- 1) SA levels ----
  for (int l = 0; l < 4; ++l) {
    int Nl = NPTS[l], np = NPOINT[l];
    size_t shmem = ((size_t)4 * Nl + 2 * 1024) * 4;
    fps_kernel<<<B, 1024, shmem, stream>>>(xyz[l], Nl, np, fpsidx);
    {
      int tp = B * np;
      gather_points_kernel<<<CEILDIV(tp, 256), 256, 0, stream>>>(
          xyz[l], fpsidx, xyz[l + 1], Nl, np, tp);
      ball_query_kernel<<<CEILDIV(tp, 4), 128, 0, stream>>>(
          xyz[l], xyz[l + 1], gi, Nl, np, RAD2[l], tp);
    }
    int Kpad = wkp[l * 3 + 0];
    int Cfeat = CSA[l];
    int Mb = np * 32;                       // per-batch GEMM rows (mult of 128)
    for (int b = 0; b < B; ++b) {
      int total = Mb * Kpad;
      sa_group_kernel<<<CEILDIV(total, 256), 256, 0, stream>>>(
          xyz[l] + (size_t)b * Nl * 3, xyz[l + 1] + (size_t)b * np * 3,
          safeat[l] + (size_t)b * Nl * Cfeat, Cfeat,
          gi + (size_t)b * np * 32, G, Kpad, total);
      launch_gemm(G, A1, l * 3 + 0, Mb);
      launch_gemm(A1, A2, l * 3 + 1, Mb);
      launch_gemm(A2, A3, l * 3 + 2, Mb);
      int c3 = SAC[l][2];                   // == Npad (mult of 64)
      int tp = np * c3;
      maxpool32_kernel<<<CEILDIV(tp, 256), 256, 0, stream>>>(
          A3, safeat[l + 1] + (size_t)b * np * c3, c3, tp);
    }
  }

  // ---- 2) FP levels ----
  for (int i = 0; i < 4; ++i) {
    int t = 3 - i;
    int N1 = NPTS[t], N2 = NPTS[t + 1];
    int C1 = CSA[t];
    int C2 = (i == 0) ? CSA[4] : 256;
    const _Float16* f2 = (i == 0) ? safeat[4] : fpout[i - 1];
    int Kpad = wkp[12 + i * 2];
    {
      int tp = B * N1;
      three_nn_kernel<<<CEILDIV(tp, 256), 256, 0, stream>>>(
          xyz[t], xyz[t + 1], nnidx, nnw, N1, N2, tp);
      int total = B * N1 * Kpad;
      fp_build_kernel<<<CEILDIV(total, 256), 256, 0, stream>>>(
          safeat[t], C1, f2, C2, nnidx, nnw, G, N1, N2, Kpad, total);
    }
    int M = B * N1;                          // 4096..32768, multiples of 128
    launch_gemm(G, A1, 12 + i * 2 + 0, M);
    launch_gemm(A1, fpout[i], 12 + i * 2 + 1, M);
  }

  // ---- 3) heads (levels 0,1,2 = fpout[3], fpout[2], fpout[1]) ----
  float* outp = (float*)d_out;
  size_t obase = 0;
  for (int hi = 0; hi < 3; ++hi) {
    int Nl = (hi == 0) ? 8192 : (hi == 1) ? 4096 : 2048;
    const _Float16* f = fpout[3 - hi];
    int base = HEADB + hi * 8;
    int total = B * Nl * 26;
    head_kernel<<<CEILDIV(total, 256), 256, 0, stream>>>(
        f,
        (const float*)d_in[base + 0], (const float*)d_in[base + 1],
        (const float*)d_in[base + 2], (const float*)d_in[base + 3],
        (const float*)d_in[base + 4], (const float*)d_in[base + 5],
        (const float*)d_in[base + 6], (const float*)d_in[base + 7],
        outp + obase, Nl, total);
    obase += (size_t)B * 26 * Nl;
  }
}
